// Model_57990648430640
// MI455X (gfx1250) — matrix operand order repaired and checked by reading, not compiled, not run on hardware
//
#include <hip/hip_runtime.h>
#include <hip/hip_bf16.h>

typedef __attribute__((ext_vector_type(8)))  _Float16 v8h;
typedef __attribute__((ext_vector_type(16))) _Float16 v16h;
typedef __attribute__((ext_vector_type(8)))  float    v8f;

#define BDIM 8192
#define DDIM 256
#define TEMP_INV   14.285714285714286f      // 1/0.07
#define EXP2_SCALE 20.609929155556670f      // log2(e)/0.07
#define KTILE 64
#define LPITCH 72                           // 64 + 8 halves pad -> 144B pitch, conflict-free

// ---------------- Kernel 1: row-normalize both inputs, emit f16 ----------------
__global__ __launch_bounds__(256) void k_normalize(const float* __restrict__ q,
                                                   const float* __restrict__ g,
                                                   _Float16* __restrict__ qh,
                                                   _Float16* __restrict__ gh) {
  const int wave = threadIdx.x >> 5;
  const int lane = threadIdx.x & 31;
  const int row  = blockIdx.x * 8 + wave;

  const float4* q4 = (const float4*)(q + (size_t)row * DDIM) + lane * 2;
  const float4* g4 = (const float4*)(g + (size_t)row * DDIM) + lane * 2;
  float4 a0 = q4[0], a1 = q4[1];
  float4 b0 = g4[0], b1 = g4[1];

  float aq[8] = {a0.x, a0.y, a0.z, a0.w, a1.x, a1.y, a1.z, a1.w};
  float ag[8] = {b0.x, b0.y, b0.z, b0.w, b1.x, b1.y, b1.z, b1.w};

  float sq = 0.f, sg = 0.f;
#pragma unroll
  for (int i = 0; i < 8; ++i) { sq += aq[i] * aq[i]; sg += ag[i] * ag[i]; }
#pragma unroll
  for (int off = 16; off; off >>= 1) {
    sq += __shfl_xor(sq, off, 32);
    sg += __shfl_xor(sg, off, 32);
  }
  const float iq = 1.0f / fmaxf(sqrtf(sq), 1e-8f);
  const float ig = 1.0f / fmaxf(sqrtf(sg), 1e-8f);

  v8h hq, hg;
#pragma unroll
  for (int i = 0; i < 8; ++i) {
    hq[i] = (_Float16)(aq[i] * iq);
    hg[i] = (_Float16)(ag[i] * ig);
  }
  *(v8h*)(qh + (size_t)row * DDIM + lane * 8) = hq;
  *(v8h*)(gh + (size_t)row * DDIM + lane * 8) = hg;
}

// ------ Kernel 2: fused S = q @ [q;g]^T -> exp -> masked row-sum partials ------
__global__ __launch_bounds__(256) void k_gemm_exp(const _Float16* __restrict__ qh,
                                                  const _Float16* __restrict__ gh,
                                                  float* __restrict__ rowpart,
                                                  float* __restrict__ diag) {
  __shared__ _Float16 As[128 * LPITCH];
  __shared__ _Float16 Bs[128 * LPITCH];
  __shared__ float    part[128][2];

  const int t      = threadIdx.x;
  const int lane   = t & 31;
  const int waveId = t >> 5;
  const int waveM  = waveId >> 1;   // 0..3
  const int waveN  = waveId & 1;    // 0..1
  const int lane15 = lane & 15;
  const bool hiHalf = lane >= 16;

  const int mBase = blockIdx.x * 128;
  const int nBase = blockIdx.y * 128;

  const _Float16* Aptr = qh + (size_t)mBase * DDIM;
  const _Float16* Bptr = (nBase < BDIM) ? (qh + (size_t)nBase * DDIM)
                                        : (gh + (size_t)(nBase - BDIM) * DDIM);

  v8f acc[2][4] = {};

  for (int kc = 0; kc < DDIM; kc += KTILE) {
    // Stage 128x64 tiles of A and K into LDS (each thread: 4x16B per tile)
#pragma unroll
    for (int i = 0; i < 4; ++i) {
      const int c   = t + i * 256;    // 0..1023
      const int row = c >> 3;         // 0..127
      const int col = (c & 7) * 8;    // 0..56
      *(v8h*)(&As[row * LPITCH + col]) = *(const v8h*)(Aptr + (size_t)row * DDIM + kc + col);
      *(v8h*)(&Bs[row * LPITCH + col]) = *(const v8h*)(Bptr + (size_t)row * DDIM + kc + col);
    }
    __syncthreads();

#pragma unroll
    for (int ks = 0; ks < 2; ++ks) {
      const int k0 = ks * 32;
      v16h a[2], b[4];

      // A fragment: lanes 0-15 -> K {0..7,16..23}, lanes 16-31 -> K {8..15,24..31}
      const int aoff = k0 + (hiHalf ? 8 : 0);
#pragma unroll
      for (int mi = 0; mi < 2; ++mi) {
        const _Float16* p = &As[(waveM * 32 + mi * 16 + lane15) * LPITCH + aoff];
        v8h lo = *(const v8h*)p;
        v8h hi = *(const v8h*)(p + 16);
        a[mi] = __builtin_shufflevector(lo, hi, 0,1,2,3,4,5,6,7,8,9,10,11,12,13,14,15);
      }
      // B fragment: lanes 0-15 -> K {0..7,16..23}, lanes 16-31 -> K {8..15,24..31} of one key row
      const int boff = k0 + (hiHalf ? 8 : 0);
#pragma unroll
      for (int ni = 0; ni < 4; ++ni) {
        const _Float16* p = &Bs[(waveN * 64 + ni * 16 + lane15) * LPITCH + boff];
        v8h lo = *(const v8h*)p;
        v8h hi = *(const v8h*)(p + 16);
        b[ni] = __builtin_shufflevector(lo, hi, 0,1,2,3,4,5,6,7,8,9,10,11,12,13,14,15);
      }

#pragma unroll
      for (int mi = 0; mi < 2; ++mi)
#pragma unroll
        for (int ni = 0; ni < 4; ++ni)
          acc[mi][ni] = __builtin_amdgcn_wmma_f32_16x16x32_f16(
              false, a[mi], false, b[ni], (short)0, acc[mi][ni], false, false);
    }
    __syncthreads();
  }

  // Epilogue: exp, diagonal handling, deterministic row-sum partials
  const int rOff = hiHalf ? 8 : 0;
#pragma unroll
  for (int mi = 0; mi < 2; ++mi) {
#pragma unroll
    for (int v = 0; v < 8; ++v) {
      const int mLocal = waveM * 32 + mi * 16 + rOff + v;
      const int m = mBase + mLocal;
      float psum = 0.f;
#pragma unroll
      for (int ni = 0; ni < 4; ++ni) {
        const int n = nBase + waveN * 64 + ni * 16 + lane15;
        const float s = acc[mi][ni][v];
        const float e = exp2f(s * EXP2_SCALE);
        if (n == m + BDIM) diag[m] = s;        // qg diagonal (log-numerator), unique writer
        psum += (n == m) ? 0.f : e;            // exclude qq diagonal from denominator
      }
#pragma unroll
      for (int off = 8; off; off >>= 1)
        psum += __shfl_xor(psum, off, 32);     // reduce within each 16-lane half
      if (lane15 == 0) part[mLocal][waveN] = psum;  // unique (row, waveN) writer
    }
  }
  __syncthreads();
  if (t < 128)
    rowpart[(size_t)blockIdx.y * BDIM + mBase + t] = part[t][0] + part[t][1];
}

// -------- Kernel 3: per-row deterministic reduction of the 128 N-partials --------
__global__ __launch_bounds__(256) void k_rowfinish(const float* __restrict__ rowpart,
                                                   const float* __restrict__ diag,
                                                   float* __restrict__ lossterm) {
  const int i = blockIdx.x * 256 + threadIdx.x;
  float s = 0.f;
  for (int p = 0; p < 128; ++p)
    s += rowpart[(size_t)p * BDIM + i];        // denom_i = pos + all negatives
  lossterm[i] = logf(s) - diag[i] * TEMP_INV;  // log(denom) - log(pos)
}

// -------- Kernel 4: scalar reduction --------
__global__ __launch_bounds__(256) void k_reduce(const float* __restrict__ lossterm,
                                                float* __restrict__ out) {
  __shared__ float red[256];
  float s = 0.f;
  for (int i = threadIdx.x; i < BDIM; i += 256) s += lossterm[i];
  red[threadIdx.x] = s;
  __syncthreads();
  for (int st = 128; st; st >>= 1) {
    if (threadIdx.x < st) red[threadIdx.x] += red[threadIdx.x + st];
    __syncthreads();
  }
  if (threadIdx.x == 0) out[0] = red[0] * (1.0f / (float)BDIM);
}

extern "C" void kernel_launch(void* const* d_in, const int* in_sizes, int n_in,
                              void* d_out, int out_size, void* d_ws, size_t ws_size,
                              hipStream_t stream) {
  const float* q = (const float*)d_in[0];
  const float* g = (const float*)d_in[1];

  char* ws = (char*)d_ws;
  _Float16* qh      = (_Float16*)ws;                      // 4 MB
  _Float16* gh      = qh + (size_t)BDIM * DDIM;           // 4 MB
  float*    diag    = (float*)(gh + (size_t)BDIM * DDIM); // 32 KB
  float*    lossterm = diag + BDIM;                       // 32 KB
  float*    rowpart  = lossterm + BDIM;                   // 4 MB (128 x 8192)

  k_normalize<<<BDIM / 8, 256, 0, stream>>>(q, g, qh, gh);
  k_gemm_exp<<<dim3(BDIM / 128, 2 * BDIM / 128), 256, 0, stream>>>(qh, gh, rowpart, diag);
  k_rowfinish<<<BDIM / 256, 256, 0, stream>>>(rowpart, diag, lossterm);
  k_reduce<<<1, 256, 0, stream>>>(lossterm, (float*)d_out);
}
